// TwinBottleneck_14654428414269
// MI455X (gfx1250) — matrix operand order repaired and checked by reading, not compiled, not run on hardware
//
#include <hip/hip_runtime.h>
#include <hip/hip_bf16.h>

// Problem constants (from reference setup_inputs).
#define N_ROWS 8192
#define BBN_D  64
#define CBN_D  512
#define STRIP  256            // adjacency strip width (j columns) staged in LDS
#define SPITCH 264            // strip row pitch (ushorts): 256 + 8 pad, keeps 16B align,
                              // spreads per-m ds_load_b128 across LDS banks
#define MTILE  32             // rows per block in the big GEMMs (2 x 16-row WMMA tiles)

typedef __attribute__((ext_vector_type(16))) unsigned short v16u;
typedef __attribute__((ext_vector_type(16))) __bf16         v16bf;
typedef __attribute__((ext_vector_type(8)))  float          v8f;

union Frag {
  v16u           v;
  uint4          q[2];
  unsigned short s[16];
};

static __device__ __forceinline__ unsigned short f2bf(float f) {
  unsigned int u = __float_as_uint(f);
  u += 0x7FFFu + ((u >> 16) & 1u);   // round-to-nearest-even
  return (unsigned short)(u >> 16);
}

static __device__ __forceinline__ v8f wmma_bf16(v16u a, v16u b, v8f c) {
  return __builtin_amdgcn_wmma_f32_16x16x32_bf16(
      false, __builtin_bit_cast(v16bf, a),
      false, __builtin_bit_cast(v16bf, b),
      (short)0, c, false, false);
}

// adjacency element from Gram accumulator g = bbn_i . bbn_j
static __device__ __forceinline__ float adj_elem(float g, float si, float sj) {
  float cc = 2.0f * g - si - sj;                 // c = 2G - s_i - s_j
  float nd = fminf(fabsf(cc) * (1.0f / 64.0f), 1.0f);
  float x  = 1.0f - nd;
  // pow(x, 1.4) = exp2(1.4 * log2(x)); v_log_f32 / v_exp_f32
  return (x > 0.0f) ? __builtin_amdgcn_exp2f(1.4f * __builtin_amdgcn_logf(x)) : 0.0f;
}

static __device__ __forceinline__ float sigmoidf_fast(float v) {
  return 1.0f / (1.0f + __builtin_amdgcn_exp2f(-v * 1.44269504f));
}

// A-fragment 16x32 bf16 (row-major source, row pitch ld elements).
// Lane L: m=L&15, h=L>>4; chunk0 = k0+8h .. +7, chunk1 = k0+8h+16 .. +7.
static __device__ __forceinline__ Frag load_a_rm(const unsigned short* base, int ld,
                                                 int row0, int k0, int lane) {
  int m = lane & 15, h = lane >> 4;
  const unsigned short* p = base + (size_t)(row0 + m) * ld + k0 + 8 * h;
  Frag f;
  f.q[0] = *reinterpret_cast<const uint4*>(p);
  f.q[1] = *reinterpret_cast<const uint4*>(p + 16);
  return f;
}

// B-fragment 32x16 bf16 from a TRANSPOSED operand base_t[col][k] (pitch ldt):
// lane L: n=L&15, h=L>>4; chunk0 = k0+8h .. +7, chunk1 = k0+8h+16 .. +7.
static __device__ __forceinline__ Frag load_b_tr(const unsigned short* base_t, int ldt,
                                                 int k0, int col0, int lane) {
  int n = lane & 15, h = lane >> 4;
  const unsigned short* p = base_t + (size_t)(col0 + n) * ldt + k0 + 8 * h;
  Frag f;
  f.q[0] = *reinterpret_cast<const uint4*>(p);
  f.q[1] = *reinterpret_cast<const uint4*>(p + 16);
  return f;
}

// ---------------- K1: row sums of bbn + f32 -> bf16 convert -------------------
__global__ void k_prep(const float* __restrict__ bbn, float* __restrict__ s,
                       unsigned short* __restrict__ bbn_bf) {
  int row  = blockIdx.x * 8 + (threadIdx.x >> 5);
  int lane = threadIdx.x & 31;
  const float* p = bbn + (size_t)row * BBN_D;
  float a = p[lane];
  float b = p[lane + 32];
  bbn_bf[(size_t)row * BBN_D + lane]      = f2bf(a);
  bbn_bf[(size_t)row * BBN_D + lane + 32] = f2bf(b);
  float v = a + b;
#pragma unroll
  for (int mask = 16; mask >= 1; mask >>= 1) v += __shfl_xor(v, mask, 32);
  if (lane == 0) s[row] = v;
}

// ---------------- K2: degrees d[i] = sum_j adj[i][j] --------------------------
// Block = 8 waves, one 16-row i-tile; wave w sweeps j in [w*1024, (w+1)*1024).
// bbn^T B-fragments are contiguous rows of bbn_bf. No atomics -> deterministic.
__global__ void k_degree(const unsigned short* __restrict__ bbn_bf,
                         const float* __restrict__ s, float* __restrict__ d) {
  __shared__ float sred[16][8];
  int w = threadIdx.x >> 5, lane = threadIdx.x & 31;
  int i0 = blockIdx.x * 16;
  int h  = lane >> 4;

  Frag a0 = load_a_rm(bbn_bf, BBN_D, i0, 0, lane);
  Frag a1 = load_a_rm(bbn_bf, BBN_D, i0, 32, lane);
  float si[8];
#pragma unroll
  for (int r = 0; r < 8; ++r) si[r] = s[i0 + r + 8 * h];

  float rs[8] = {0.f, 0.f, 0.f, 0.f, 0.f, 0.f, 0.f, 0.f};
  for (int t = 0; t < 64; ++t) {
    int j = w * 1024 + t * 16;
    Frag b0 = load_b_tr(bbn_bf, BBN_D, 0, j, lane);    // B = bbn^T
    Frag b1 = load_b_tr(bbn_bf, BBN_D, 32, j, lane);
    v8f acc = {};
    acc = wmma_bf16(a0.v, b0.v, acc);
    acc = wmma_bf16(a1.v, b1.v, acc);
    float sj = s[j + (lane & 15)];
#pragma unroll
    for (int r = 0; r < 8; ++r) rs[r] += adj_elem(acc[r], si[r], sj);
  }
#pragma unroll
  for (int mask = 8; mask >= 1; mask >>= 1)
#pragma unroll
    for (int r = 0; r < 8; ++r) rs[r] += __shfl_xor(rs[r], mask, 32);
  if ((lane & 15) == 0)
#pragma unroll
    for (int r = 0; r < 8; ++r) sred[r + 8 * h][w] = rs[r];
  __syncthreads();
  if (threadIdx.x < 16) {
    float acc = 0.f;
#pragma unroll
    for (int ww = 0; ww < 8; ++ww) acc += sred[threadIdx.x][ww];
    d[i0 + threadIdx.x] = acc;
  }
}

// ---------------- K3a: cbn_t[col][row] = d[row]^-1/2 * cbn[row][col] (bf16) ---
// LDS-tiled 32x32 transpose: coalesced global reads and writes.
__global__ void k_scale_cbn_t(const float* __restrict__ cbn, const float* __restrict__ d,
                              unsigned short* __restrict__ cbn_t) {
  __shared__ unsigned short tile[32][33];
  int t  = threadIdx.x;
  int r0 = blockIdx.x * 32, c0 = blockIdx.y * 32;
  int sub = t >> 5, lo = t & 31;
#pragma unroll
  for (int k = 0; k < 4; ++k) {
    int r = sub + 8 * k;
    float dis = __builtin_amdgcn_rsqf(d[r0 + r]);
    tile[lo][r] = f2bf(dis * cbn[(size_t)(r0 + r) * CBN_D + c0 + lo]);
  }
  __syncthreads();
#pragma unroll
  for (int k = 0; k < 4; ++k) {
    int c = sub + 8 * k;
    cbn_t[(size_t)(c0 + c) * N_ROWS + r0 + lo] = tile[c][lo];
  }
}

// ---------------- K3b: W_t[col][k] = bf16(W[k][col]) --------------------------
__global__ void k_convert_w_t(const float* __restrict__ W, unsigned short* __restrict__ W_t) {
  __shared__ unsigned short tile[32][33];
  int t  = threadIdx.x;
  int r0 = blockIdx.x * 32, c0 = blockIdx.y * 32;
  int sub = t >> 5, lo = t & 31;
#pragma unroll
  for (int k = 0; k < 4; ++k) {
    int r = sub + 8 * k;
    tile[lo][r] = f2bf(W[(size_t)(r0 + r) * CBN_D + c0 + lo]);
  }
  __syncthreads();
#pragma unroll
  for (int k = 0; k < 4; ++k) {
    int c = sub + 8 * k;
    W_t[(size_t)(c0 + c) * CBN_D + r0 + lo] = tile[c][lo];
  }
}

// ---------------- K4: h = D^-1/2 A D^-1/2 cbn, fused adjacency recompute ------
// Block = 8 waves, one 32-row i-block (two 16-row WMMA tiles); wave w owns 64
// output columns for BOTH row tiles (B-fragments reused -> 8 WMMAs / 12 loads).
// Per 256-wide j strip: every wave rebuilds its own 32x32 adjacency chunk
// (8 WMMAs + pow, bbn^T B-frags reused across row tiles) into a shared 32x256
// LDS strip; after one barrier the strip feeds 8 inner steps.
__global__ void k_spectral(const unsigned short* __restrict__ bbn_bf,
                           const unsigned short* __restrict__ cbn_t,
                           const float* __restrict__ s, const float* __restrict__ d,
                           unsigned short* __restrict__ h_bf) {
  __shared__ __align__(16) unsigned short adj_lds[MTILE * SPITCH];  // ~16.9 KB strip
  int w = threadIdx.x >> 5, lane = threadIdx.x & 31;
  int i0 = blockIdx.x * MTILE;
  int m = lane & 15, h = lane >> 4;
  int c0 = w * 64;

  Frag a0[2], a1[2];
  float si[2][8];
#pragma unroll
  for (int rt = 0; rt < 2; ++rt) {
    a0[rt] = load_a_rm(bbn_bf, BBN_D, i0 + 16 * rt, 0, lane);
    a1[rt] = load_a_rm(bbn_bf, BBN_D, i0 + 16 * rt, 32, lane);
#pragma unroll
    for (int r = 0; r < 8; ++r) si[rt][r] = s[i0 + 16 * rt + r + 8 * h];
  }

  v8f acc[2][4] = {{{}, {}, {}, {}}, {{}, {}, {}, {}}};
  for (int jb = 0; jb < N_ROWS; jb += STRIP) {
    __syncthreads();                      // consumers of previous strip done
    // each wave produces adjacency columns [jb + w*32, jb + w*32 + 32), both row tiles
#pragma unroll
    for (int t2 = 0; t2 < 2; ++t2) {
      int j1 = jb + w * 32 + t2 * 16;
      Frag b0 = load_b_tr(bbn_bf, BBN_D, 0, j1, lane);
      Frag b1 = load_b_tr(bbn_bf, BBN_D, 32, j1, lane);
      float sj = s[j1 + m];
#pragma unroll
      for (int rt = 0; rt < 2; ++rt) {
        v8f g = {};
        g = wmma_bf16(a0[rt].v, b0.v, g);
        g = wmma_bf16(a1[rt].v, b1.v, g);
#pragma unroll
        for (int r = 0; r < 8; ++r)
          adj_lds[(16 * rt + r + 8 * h) * SPITCH + w * 32 + t2 * 16 + m] =
              f2bf(adj_elem(g[r], si[rt][r], sj));
      }
    }
    __syncthreads();                      // strip visible
#pragma unroll
    for (int ji = 0; ji < 8; ++ji) {
      int j = jb + ji * 32;
      Frag af[2];                         // A-frags 16x32 from LDS strip
#pragma unroll
      for (int rt = 0; rt < 2; ++rt) {
        const unsigned short* lp = &adj_lds[(16 * rt + m) * SPITCH + ji * 32 + 8 * h];
        af[rt].q[0] = *reinterpret_cast<const uint4*>(lp);
        af[rt].q[1] = *reinterpret_cast<const uint4*>(lp + 16);
      }
#pragma unroll
      for (int t = 0; t < 4; ++t) {
        Frag bfr = load_b_tr(cbn_t, N_ROWS, j, c0 + 16 * t, lane);
        acc[0][t] = wmma_bf16(af[0].v, bfr.v, acc[0][t]);
        acc[1][t] = wmma_bf16(af[1].v, bfr.v, acc[1][t]);
      }
    }
  }
  // epilogue: scale rows by d_i^-1/2, store bf16
#pragma unroll
  for (int rt = 0; rt < 2; ++rt)
#pragma unroll
    for (int r = 0; r < 8; ++r) {
      int row = i0 + 16 * rt + r + 8 * h;
      float disv = __builtin_amdgcn_rsqf(d[row]);
#pragma unroll
      for (int t = 0; t < 4; ++t)
        h_bf[(size_t)row * CBN_D + c0 + 16 * t + m] = f2bf(disv * acc[rt][t][r]);
    }
}

// ---------------- K5: out = sigmoid(h @ W + b) --------------------------------
// Same 32-row M-blocking; W_t B-fragments reused across both row tiles.
__global__ void k_out(const unsigned short* __restrict__ h_bf,
                      const unsigned short* __restrict__ W_t,
                      const float* __restrict__ b, float* __restrict__ out) {
  int w = threadIdx.x >> 5, lane = threadIdx.x & 31;
  int i0 = blockIdx.x * MTILE;
  int m = lane & 15, h = lane >> 4;
  int c0 = w * 64;
  v8f acc[2][4] = {{{}, {}, {}, {}}, {{}, {}, {}, {}}};
  for (int k = 0; k < CBN_D; k += 32) {
    Frag af0 = load_a_rm(h_bf, CBN_D, i0, k, lane);
    Frag af1 = load_a_rm(h_bf, CBN_D, i0 + 16, k, lane);
#pragma unroll
    for (int t = 0; t < 4; ++t) {
      Frag bfr = load_b_tr(W_t, CBN_D, k, c0 + 16 * t, lane);
      acc[0][t] = wmma_bf16(af0.v, bfr.v, acc[0][t]);
      acc[1][t] = wmma_bf16(af1.v, bfr.v, acc[1][t]);
    }
  }
#pragma unroll
  for (int rt = 0; rt < 2; ++rt)
#pragma unroll
    for (int r = 0; r < 8; ++r) {
      int row = i0 + 16 * rt + r + 8 * h;
#pragma unroll
      for (int t = 0; t < 4; ++t) {
        int col = c0 + 16 * t + m;
        float v = acc[rt][t][r] + b[col];
        out[(size_t)row * CBN_D + col] = sigmoidf_fast(v);
      }
    }
}

extern "C" void kernel_launch(void* const* d_in, const int* in_sizes, int n_in,
                              void* d_out, int out_size, void* d_ws, size_t ws_size,
                              hipStream_t stream) {
  const float* bbn = (const float*)d_in[0];   // [8192, 64]
  const float* cbn = (const float*)d_in[1];   // [8192, 512]
  const float* W   = (const float*)d_in[2];   // [512, 512]
  const float* b   = (const float*)d_in[3];   // [512]

  char* ws = (char*)d_ws;                       // ~17.6 MB used
  float*          s      = (float*)(ws);                       // 32 KB
  float*          d      = (float*)(ws + 32768);               // 32 KB
  unsigned short* bbn_bf = (unsigned short*)(ws + 65536);      // 1 MB
  unsigned short* cbn_t  = (unsigned short*)(ws + 1114112);    // 8 MB [512][8192]
  unsigned short* W_t    = (unsigned short*)(ws + 9502720);    // 0.5 MB [512][512]
  unsigned short* h_bf   = (unsigned short*)(ws + 10027008);   // 8 MB

  k_prep       <<<N_ROWS / 8, 256, 0, stream>>>(bbn, s, bbn_bf);
  k_degree     <<<N_ROWS / 16, 256, 0, stream>>>(bbn_bf, s, d);
  k_scale_cbn_t<<<dim3(N_ROWS / 32, CBN_D / 32), 256, 0, stream>>>(cbn, d, cbn_t);
  k_convert_w_t<<<dim3(CBN_D / 32, CBN_D / 32), 256, 0, stream>>>(W, W_t);
  k_spectral   <<<N_ROWS / MTILE, 256, 0, stream>>>(bbn_bf, cbn_t, s, d, h_bf);
  k_out        <<<N_ROWS / MTILE, 256, 0, stream>>>(h_bf, W_t, b, (float*)d_out);
}
